// GCN_28226525069446
// MI455X (gfx1250) — matrix operand order repaired and checked by reading, not compiled, not run on hardware
//
#include <hip/hip_runtime.h>
#include <hip/hip_bf16.h>

#define NFEAT 256
#define NHID  128
#define KPAD  264   // 256 + 8 bf16 pad: lane stride 528B = 132 dwords -> bank step 4, conflict-free

typedef __attribute__((ext_vector_type(16))) __bf16 v16bf;
typedef __attribute__((ext_vector_type(8)))  __bf16 v8bf;
typedef __attribute__((ext_vector_type(8)))  float  v8f;

__device__ __forceinline__ __bf16 f2bf(float f) {
  unsigned u = __builtin_bit_cast(unsigned, f);
  u += 0x7FFFu + ((u >> 16) & 1u);                 // round-to-nearest-even
  unsigned short hs = (unsigned short)(u >> 16);
  return __builtin_bit_cast(__bf16, hs);
}

// ---- degree / norm ---------------------------------------------------------
__global__ void __launch_bounds__(256) gcn_deg_init(float* deg, int N) {
  int i = blockIdx.x * blockDim.x + threadIdx.x;
  if (i < N) deg[i] = 1.0f;                        // self-loop
}

__global__ void __launch_bounds__(256) gcn_deg_accum(const int* __restrict__ dst,
                                                     float* deg, int E) {
  int e = blockIdx.x * blockDim.x + threadIdx.x;
  if (e < E) atomicAdd(&deg[dst[e]], 1.0f);
}

__global__ void __launch_bounds__(256) gcn_deg_rsqrt(float* deg, int N) {
  int i = blockIdx.x * blockDim.x + threadIdx.x;
  if (i < N) deg[i] = rsqrtf(deg[i]);              // deg >= 1 guaranteed
}

// ---- h = x @ W via V_WMMA_F32_16X16X32_BF16 --------------------------------
// Block = 256 threads = 8 waves. Block owns a 16-row strip of x; wave w owns
// output columns [16w, 16w+16). x strip and full W are staged in LDS as bf16
// (W transposed so each lane reads K-contiguous data for its column).
// Tile origin is clamped so every tile is fully in-bounds: overlapping tiles
// recompute identical values -> deterministic, and no per-element store guards.
__global__ void __launch_bounds__(256)
gcn_gemm_wmma(const float* __restrict__ x, const float* __restrict__ W,
              float* __restrict__ h, int N) {
  __shared__ __bf16 xlds[16 * KPAD];
  __shared__ __bf16 wlds[NHID * KPAD];

  int row0 = blockIdx.x * 16;
  if (row0 > N - 16) row0 = N - 16;                // N >= 16 always here
  const int lane = threadIdx.x & 31;
  const int wv   = threadIdx.x >> 5;
  const int col0 = wv * 16;
  const int half = lane >> 4;     // 0: lanes 0-15, 1: lanes 16-31
  const int lid  = lane & 15;

  // stage x[row0:row0+16, :] -> bf16 (coalesced global reads)
  for (int i = threadIdx.x; i < 16 * NFEAT; i += 256) {
    int r = i >> 8, k = i & (NFEAT - 1);
    xlds[r * KPAD + k] = f2bf(x[(size_t)(row0 + r) * NFEAT + k]);
  }
  // stage W (256x128) transposed -> wlds[n*KPAD + k] (coalesced global reads)
  for (int i = threadIdx.x; i < NFEAT * NHID; i += 256) {
    int k = i >> 7, n = i & (NHID - 1);
    wlds[n * KPAD + k] = f2bf(W[i]);
  }
  __syncthreads();

  v8f acc = {};
  #pragma unroll
  for (int kb = 0; kb < NFEAT; kb += 32) {
    // A 16x32 bf16 frag (ISA 7.12.2): lane half h, row lid:
    //   elems 0..7  = K kb+8h .. kb+8h+7
    //   elems 8..15 = K kb+16+8h .. kb+16+8h+7
    v8bf alo = *(const v8bf*)&xlds[lid * KPAD + kb + 8 * half];
    v8bf ahi = *(const v8bf*)&xlds[lid * KPAD + kb + 16 + 8 * half];
    // B 32x16 bf16 frag: lane holds column col0+lid, elems 0..7 = K kb+8h..kb+8h+7, elems 8..15 = K kb+16+8h..kb+16+8h+7
    v8bf blo = *(const v8bf*)&wlds[(col0 + lid) * KPAD + kb + 8 * half];
    v8bf bhi = *(const v8bf*)&wlds[(col0 + lid) * KPAD + kb + 16 + 8 * half];
    v16bf a = __builtin_shufflevector(alo, ahi, 0,1,2,3,4,5,6,7,8,9,10,11,12,13,14,15);
    v16bf bm = __builtin_shufflevector(blo, bhi, 0,1,2,3,4,5,6,7,8,9,10,11,12,13,14,15);
    acc = __builtin_amdgcn_wmma_f32_16x16x32_bf16(
        /*neg_a=*/false, a, /*neg_b=*/false, bm,
        /*c_mod=*/(short)0, acc, /*reuse_a=*/false, /*reuse_b=*/false);
  }

  // C layout: elem v -> M = v + 8*half, N = lid (no guards: tile is in-bounds)
  float* hp = h + (size_t)(row0 + 8 * half) * NHID + col0 + lid;
  #pragma unroll
  for (int v = 0; v < 8; ++v)
    hp[(size_t)v * NHID] = acc[v];
}

// ---- out[i] = h[i] * dinv[i]^2 + b (self-loop term, also initializes out) --
__global__ void __launch_bounds__(256)
gcn_out_init(const float* __restrict__ h, const float* __restrict__ dinv,
             const float* __restrict__ b, float* __restrict__ out, int N) {
  int t = blockIdx.x * blockDim.x + threadIdx.x;
  int node = t >> 5;
  int f = (t & 31) * 4;
  if (node >= N) return;
  float di = dinv[node];
  float sn = di * di;
  const float4 hv = *(const float4*)(h + (size_t)node * NHID + f);
  const float4 bv = *(const float4*)(b + f);
  float4 o;
  o.x = hv.x * sn + bv.x;
  o.y = hv.y * sn + bv.y;
  o.z = hv.z * sn + bv.z;
  o.w = hv.w * sn + bv.w;
  *(float4*)(out + (size_t)node * NHID + f) = o;
}

// ---- edge scatter: one wave per edge, float4 per lane, L2-resident atomics -
__global__ void __launch_bounds__(256)
gcn_scatter(const int* __restrict__ src, const int* __restrict__ dst,
            const float* __restrict__ h, const float* __restrict__ dinv,
            float* __restrict__ out, int E) {
  const int lane = threadIdx.x & 31;
  const int wpb  = blockDim.x >> 5;
  int e = blockIdx.x * wpb + (threadIdx.x >> 5);
  const int stride = gridDim.x * wpb;
  for (; e < E; e += stride) {
    const int s = src[e];
    const int d = dst[e];
    const float norm = dinv[s] * dinv[d];
    const float4 hv = *(const float4*)(h + (size_t)s * NHID + lane * 4);
    float* op = out + (size_t)d * NHID + lane * 4;
    atomicAdd(op + 0, hv.x * norm);
    atomicAdd(op + 1, hv.y * norm);
    atomicAdd(op + 2, hv.z * norm);
    atomicAdd(op + 3, hv.w * norm);
  }
}

extern "C" void kernel_launch(void* const* d_in, const int* in_sizes, int n_in,
                              void* d_out, int out_size, void* d_ws, size_t ws_size,
                              hipStream_t stream) {
  (void)n_in; (void)out_size; (void)ws_size;
  const float* x = (const float*)d_in[0];
  const float* W = (const float*)d_in[1];
  const float* b = (const float*)d_in[2];
  const int*   ei = (const int*)d_in[3];   // edge_index flat [2, E]
  const int N = in_sizes[0] / NFEAT;
  const int E = in_sizes[3] / 2;
  const int* src = ei;
  const int* dst = ei + E;
  float* out = (float*)d_out;

  // workspace: [deg/dinv : N floats (aligned)] [h : N*NHID floats]
  float* deg = (float*)d_ws;
  float* h   = deg + (((size_t)N + 255) & ~(size_t)255);

  gcn_deg_init  <<<(N + 255) / 256, 256, 0, stream>>>(deg, N);
  gcn_deg_accum <<<(E + 255) / 256, 256, 0, stream>>>(dst, deg, E);
  gcn_deg_rsqrt <<<(N + 255) / 256, 256, 0, stream>>>(deg, N);   // deg -> dinv in place
  gcn_gemm_wmma <<<(N + 15) / 16, 256, 0, stream>>>(x, W, h, N);
  gcn_out_init  <<<((unsigned)N * 32 + 255) / 256, 256, 0, stream>>>(h, deg, b, out, N);
  gcn_scatter   <<<(E + 7) / 8, 256, 0, stream>>>(src, dst, h, deg, out, E);
}
